// SimpleAttention_10428180594842
// MI455X (gfx1250) — compile-verified, matrix operand order checked by reading, not run on hardware
//
#include <hip/hip_runtime.h>
#include <hip/hip_bf16.h>

typedef __attribute__((ext_vector_type(16))) _Float16 v16h;
typedef __attribute__((ext_vector_type(8)))  float    v8f;

#define B_  4
#define N_  2048
#define D_  1024
#define H_  8
#define DH_ 64
#define INNER_ 512
#define ROWS_ (B_ * N_)   // 8192

#define WMMA_F16(a, b, c) \
  __builtin_amdgcn_wmma_f32_16x16x32_f16(false, (a), false, (b), (short)0, (c), false, false)

__device__ __forceinline__ float redmax16(float v) {
  #pragma unroll
  for (int m = 1; m < 16; m <<= 1) v = fmaxf(v, __shfl_xor(v, m, 32));
  return v;
}
__device__ __forceinline__ float redsum16(float v) {
  #pragma unroll
  for (int m = 1; m < 16; m <<= 1) v += __shfl_xor(v, m, 32);
  return v;
}
__device__ __forceinline__ float redsum32(float v) {
  #pragma unroll
  for (int m = 1; m < 32; m <<= 1) v += __shfl_xor(v, m, 32);
  return v;
}

// async copy of 32 bytes/lane: global -> LDS (tracked by ASYNCcnt).
// INST_OFFSET applies to both LDS and global addresses (ISA 08 §4.4).
__device__ __forceinline__ void async_cp32(const _Float16* g, _Float16* l) {
  unsigned loff = (unsigned)(uintptr_t)l;   // low 32 bits of generic ptr = LDS offset
  asm volatile("global_load_async_to_lds_b128 %0, %1, off\n\t"
               "global_load_async_to_lds_b128 %0, %1, off offset:16"
               :: "v"(loff), "v"(g) : "memory");
}
__device__ __forceinline__ void wait_async0() {
  asm volatile("s_wait_asynccnt 0x0" ::: "memory");
}

// ---- transpose + f32->f16 weight conversion: Wt[n*K + k] = W[k*N + n] ----
__global__ void wt_kernel(const float* __restrict__ W, _Float16* __restrict__ Wt,
                          int K, int N) {
  int idx = blockIdx.x * 256 + threadIdx.x;
  if (idx < K * N) {
    int n = idx / K, k = idx % K;
    Wt[idx] = (_Float16)W[k * N + n];
  }
}

// ---- LayerNorm rows of f32 [rows x 1024] -> f16 out; 1 wave / row ----
__global__ void ln_to_h_kernel(const float* __restrict__ x, const float* __restrict__ g,
                               _Float16* __restrict__ out) {
  int lane = threadIdx.x & 31, wid = threadIdx.x >> 5;
  int row = blockIdx.x * 4 + wid;
  const float* xr = x + (size_t)row * D_;
  float v[32], sum = 0.f, sq = 0.f;
  #pragma unroll
  for (int j = 0; j < 32; ++j) {
    v[j] = xr[j * 32 + lane];
    sum += v[j]; sq += v[j] * v[j];
  }
  sum = redsum32(sum); sq = redsum32(sq);
  float mean = sum * (1.0f / D_);
  float var  = sq * (1.0f / D_) - mean * mean;
  float rs   = rsqrtf(var + 1e-5f);
  _Float16* o = out + (size_t)row * D_;
  #pragma unroll
  for (int j = 0; j < 32; ++j)
    o[j * 32 + lane] = (_Float16)((v[j] - mean) * rs * g[j * 32 + lane]);
}

// ---- in-place LayerNorm f32 [rows x 1024] ----
__global__ void ln_inplace_kernel(float* __restrict__ x, const float* __restrict__ g) {
  int lane = threadIdx.x & 31, wid = threadIdx.x >> 5;
  int row = blockIdx.x * 4 + wid;
  float* xr = x + (size_t)row * D_;
  float v[32], sum = 0.f, sq = 0.f;
  #pragma unroll
  for (int j = 0; j < 32; ++j) {
    v[j] = xr[j * 32 + lane];
    sum += v[j]; sq += v[j] * v[j];
  }
  sum = redsum32(sum); sq = redsum32(sq);
  float mean = sum * (1.0f / D_);
  float var  = sq * (1.0f / D_) - mean * mean;
  float rs   = rsqrtf(var + 1e-5f);
  #pragma unroll
  for (int j = 0; j < 32; ++j)
    xr[j * 32 + lane] = (v[j] - mean) * rs * g[j * 32 + lane];
}

// ---- WMMA GEMM: C[MxN](f32) = A[MxK](f16 row-major) * Bt[NxK](f16 row-major)^T
// grid = (N/64, M/32), block = 32. 32x64 tile per wave, 8 accumulators,
// software-pipelined (next k-step's 6 operand loads overlap current 8 WMMAs).
__global__ void gemm_kernel(const _Float16* __restrict__ A, const _Float16* __restrict__ Bt,
                            float* __restrict__ C, int M, int N, int K) {
  int lane = threadIdx.x & 31;
  int half = lane >> 4, lcol = lane & 15;
  int m0 = blockIdx.y * 32, n0 = blockIdx.x * 64;
  const _Float16* a0p = A + (size_t)(m0 + lcol) * K + half * 16;
  const _Float16* a1p = A + (size_t)(m0 + 16 + lcol) * K + half * 16;
  const _Float16* b0p = Bt + (size_t)(n0 +  0 + lcol) * K + half * 16;
  const _Float16* b1p = Bt + (size_t)(n0 + 16 + lcol) * K + half * 16;
  const _Float16* b2p = Bt + (size_t)(n0 + 32 + lcol) * K + half * 16;
  const _Float16* b3p = Bt + (size_t)(n0 + 48 + lcol) * K + half * 16;

  v8f c00 = {}, c01 = {}, c02 = {}, c03 = {};
  v8f c10 = {}, c11 = {}, c12 = {}, c13 = {};

  v16h a0 = *(const v16h*)a0p, a1 = *(const v16h*)a1p;
  v16h b0 = *(const v16h*)b0p, b1 = *(const v16h*)b1p;
  v16h b2 = *(const v16h*)b2p, b3 = *(const v16h*)b3p;

  for (int k = 32;; k += 32) {
    int kn = (k < K) ? k : 0;                 // clamp last prefetch in-bounds
    v16h A0n = *(const v16h*)(a0p + kn);
    v16h A1n = *(const v16h*)(a1p + kn);
    v16h B0n = *(const v16h*)(b0p + kn);
    v16h B1n = *(const v16h*)(b1p + kn);
    v16h B2n = *(const v16h*)(b2p + kn);
    v16h B3n = *(const v16h*)(b3p + kn);

    c00 = WMMA_F16(a0, b0, c00);
    c01 = WMMA_F16(a0, b1, c01);
    c02 = WMMA_F16(a0, b2, c02);
    c03 = WMMA_F16(a0, b3, c03);
    c10 = WMMA_F16(a1, b0, c10);
    c11 = WMMA_F16(a1, b1, c11);
    c12 = WMMA_F16(a1, b2, c12);
    c13 = WMMA_F16(a1, b3, c13);

    if (k >= K) break;
    a0 = A0n; a1 = A1n; b0 = B0n; b1 = B1n; b2 = B2n; b3 = B3n;
  }

  v8f row0[4] = {c00, c01, c02, c03};
  v8f row1[4] = {c10, c11, c12, c13};
  #pragma unroll
  for (int t = 0; t < 4; ++t)
    #pragma unroll
    for (int r = 0; r < 8; ++r) {
      C[(size_t)(m0 + r + half * 8) * N + n0 + t * 16 + lcol]      = row0[t][r];
      C[(size_t)(m0 + 16 + r + half * 8) * N + n0 + t * 16 + lcol] = row1[t][r];
    }
}

// ---- q l2norm: q_f32[8192][512] -> qn[b][h][n][64] f16, net scale 64/max(16*||q||,1e-12)
__global__ void norm_q_kernel(const float* __restrict__ q, _Float16* __restrict__ qn) {
  int lane = threadIdx.x & 31, wid = threadIdx.x >> 5;
  int u  = blockIdx.x * 4 + wid;          // 0 .. 65535
  int bn = u >> 3, h = u & 7;
  const float* qr = q + (size_t)bn * INNER_ + h * DH_;
  float x0 = qr[lane * 2], x1 = qr[lane * 2 + 1];
  float sq = redsum32(x0 * x0 + x1 * x1);
  float scale = 64.0f / fmaxf(16.0f * sqrtf(sq), 1e-12f);
  int b = bn >> 11, n = bn & 2047;
  _Float16* dst = qn + (size_t)(((b * H_ + h) * N_) + n) * DH_;
  dst[lane * 2]     = (_Float16)(x0 * scale);
  dst[lane * 2 + 1] = (_Float16)(x1 * scale);
}

// ---- k l2norm (*4) -> kn[b][n][64]; v -> vt[b][d][n] (transposed, f16) ----
__global__ void norm_kv_kernel(const float* __restrict__ kv, _Float16* __restrict__ kn,
                               _Float16* __restrict__ vt) {
  int lane = threadIdx.x & 31, wid = threadIdx.x >> 5;
  int row = blockIdx.x * 4 + wid;          // 0 .. 8191
  const float* kr = kv + (size_t)row * 128;
  float k0 = kr[lane * 2], k1 = kr[lane * 2 + 1];
  float sq = redsum32(k0 * k0 + k1 * k1);
  float scale = 4.0f / fmaxf(sqrtf(sq), 1e-12f);
  int b = row >> 11, n = row & 2047;
  _Float16* kd = kn + (size_t)(b * N_ + n) * DH_;
  kd[lane * 2]     = (_Float16)(k0 * scale);
  kd[lane * 2 + 1] = (_Float16)(k1 * scale);
  _Float16* vb = vt + (size_t)b * DH_ * N_;
  vb[(size_t)(lane * 2)     * N_ + n] = (_Float16)kr[64 + lane * 2];
  vb[(size_t)(lane * 2 + 1) * N_ + n] = (_Float16)kr[64 + lane * 2 + 1];
}

// ---- attention: per (b,h,16-row block), 4 row-blocks per 128-thread block.
// K and V chunks (32 j x 64 d) are staged cooperatively into LDS via
// async global->LDS copies (ASYNCcnt), double-buffered; WMMA B-operands
// then come from contiguous 32B ds loads. 2-pass softmax.
__global__ void attn_kernel(const _Float16* __restrict__ qn, const _Float16* __restrict__ kn,
                            const _Float16* __restrict__ vt, _Float16* __restrict__ ao) {
  __shared__ _Float16 Kbuf[2][32 * 64];   // [chunk rows j][d]      8 KB
  __shared__ _Float16 Vbuf[2][64 * 32];   // [d][chunk cols j]      8 KB
  __shared__ _Float16 ldsP[4][16 * 32];   // per-wave P staging     4 KB

  int tid  = threadIdx.x;
  int lane = tid & 31, wid = tid >> 5;
  int half = lane >> 4, lcol = lane & 15;
  int unit = blockIdx.x * 4 + wid;         // 0 .. 4095
  int ib = unit >> 10;
  int ih = (unit >> 7) & 7;
  int i0 = (unit & 127) * 16;
  const _Float16* Q  = qn + (size_t)((ib * H_ + ih) * N_) * DH_;
  const _Float16* Kp = kn + (size_t)(ib * N_) * DH_;
  const _Float16* V  = vt + (size_t)ib * DH_ * N_;
  int d0 = half * 16;

  v16h a0 = *(const v16h*)(Q + (size_t)(i0 + lcol) * DH_ + d0);
  v16h a1 = *(const v16h*)(Q + (size_t)(i0 + lcol) * DH_ + 32 + d0);

  // ---- pass 1: row max; K chunks async-staged through LDS ----
  float m[8];
  #pragma unroll
  for (int r = 0; r < 8; ++r) m[r] = -1e30f;

  // chunk = 32 j-rows (4 KB). 128 threads x 32B = one chunk.
  async_cp32(Kp + tid * 16, &Kbuf[0][tid * 16]);       // stage chunk 0
  for (int c = 0; c < 64; ++c) {
    int cb = c & 1;
    wait_async0();
    __syncthreads();           // chunk cb ready; buffer (c+1)&1 free
    if (c < 63)
      async_cp32(Kp + (size_t)(c + 1) * 32 * DH_ + tid * 16,
                 &Kbuf[(c + 1) & 1][tid * 16]);
    #pragma unroll
    for (int sub = 0; sub < 2; ++sub) {
      const _Float16* kb = &Kbuf[cb][(sub * 16 + lcol) * DH_];
      v16h b0 = *(const v16h*)(kb + d0);
      v16h b1 = *(const v16h*)(kb + 32 + d0);
      v8f acc = {};
      acc = WMMA_F16(a0, b0, acc);
      acc = WMMA_F16(a1, b1, acc);
      #pragma unroll
      for (int r = 0; r < 8; ++r) m[r] = fmaxf(m[r], acc[r]);
    }
  }
  #pragma unroll
  for (int r = 0; r < 8; ++r) m[r] = redmax16(m[r]);

  // ---- pass 2: exp, stage P via LDS, P@V; K and V chunks async-staged ----
  float s[8];
  #pragma unroll
  for (int r = 0; r < 8; ++r) s[r] = 0.f;
  v8f o0 = {}, o1 = {}, o2 = {}, o3 = {};
  _Float16* P = ldsP[wid];
  int vd = tid >> 1, vpart = tid & 1;      // V staging role of this thread

  async_cp32(Kp + tid * 16, &Kbuf[0][tid * 16]);
  async_cp32(V + (size_t)vd * N_ + vpart * 16, &Vbuf[0][vd * 32 + vpart * 16]);

  for (int c = 0; c < 64; ++c) {
    int cb = c & 1;
    wait_async0();
    __syncthreads();           // K/V chunk cb ready; buffer (c+1)&1 free
    if (c < 63) {
      int nb = (c + 1) & 1;
      async_cp32(Kp + (size_t)(c + 1) * 32 * DH_ + tid * 16, &Kbuf[nb][tid * 16]);
      async_cp32(V + (size_t)vd * N_ + (c + 1) * 32 + vpart * 16,
                 &Vbuf[nb][vd * 32 + vpart * 16]);
    }
    #pragma unroll
    for (int sub = 0; sub < 2; ++sub) {
      const _Float16* kb = &Kbuf[cb][(sub * 16 + lcol) * DH_];
      v16h b0 = *(const v16h*)(kb + d0);
      v16h b1 = *(const v16h*)(kb + 32 + d0);
      v8f acc = {};
      acc = WMMA_F16(a0, b0, acc);
      acc = WMMA_F16(a1, b1, acc);
      #pragma unroll
      for (int r = 0; r < 8; ++r) {
        float p = __expf(acc[r] - m[r]);
        s[r] += p;
        P[(r + half * 8) * 32 + sub * 16 + lcol] = (_Float16)p;
      }
    }
    __syncthreads();           // P visible (block-uniform control flow)
    v16h ap  = *(const v16h*)(P + lcol * 32 + half * 16);
    v16h bv0 = *(const v16h*)(&Vbuf[cb][( 0 + lcol) * 32 + half * 16]);
    v16h bv1 = *(const v16h*)(&Vbuf[cb][(16 + lcol) * 32 + half * 16]);
    v16h bv2 = *(const v16h*)(&Vbuf[cb][(32 + lcol) * 32 + half * 16]);
    v16h bv3 = *(const v16h*)(&Vbuf[cb][(48 + lcol) * 32 + half * 16]);
    o0 = WMMA_F16(ap, bv0, o0);
    o1 = WMMA_F16(ap, bv1, o1);
    o2 = WMMA_F16(ap, bv2, o2);
    o3 = WMMA_F16(ap, bv3, o3);
  }

  #pragma unroll
  for (int r = 0; r < 8; ++r) s[r] = redsum16(s[r]);
  v8f os[4] = {o0, o1, o2, o3};
  #pragma unroll
  for (int r = 0; r < 8; ++r) {
    float inv = 1.0f / fmaxf(s[r], 1e-20f);
    int row = ib * N_ + i0 + r + half * 8;
    _Float16* dst = ao + (size_t)row * INNER_ + ih * DH_;
    #pragma unroll
    for (int t = 0; t < 4; ++t)
      dst[t * 16 + lcol] = (_Float16)(os[t][r] * inv);
  }
}

extern "C" void kernel_launch(void* const* d_in, const int* in_sizes, int n_in,
                              void* d_out, int out_size, void* d_ws, size_t ws_size,
                              hipStream_t stream) {
  const float* x          = (const float*)d_in[0];
  const float* norm_g     = (const float*)d_in[1];
  const float* Wq         = (const float*)d_in[2];
  const float* Wkv        = (const float*)d_in[3];
  const float* Wout       = (const float*)d_in[4];
  const float* out_norm_g = (const float*)d_in[5];
  float* out = (float*)d_out;

  char* w = (char*)d_ws;
  size_t off = 0;
  auto alloc = [&](size_t bytes) -> void* {
    void* p = w + off;
    off += (bytes + 255) & ~(size_t)255;
    return p;
  };
  _Float16* xn_h  = (_Float16*)alloc((size_t)ROWS_ * D_ * 2);       // 16 MB
  _Float16* WqT   = (_Float16*)alloc((size_t)INNER_ * D_ * 2);      //  1 MB
  _Float16* WkvT  = (_Float16*)alloc((size_t)128 * D_ * 2);         // .25 MB
  _Float16* WoutT = (_Float16*)alloc((size_t)D_ * INNER_ * 2);      //  1 MB
  float*    q_f32 = (float*)   alloc((size_t)ROWS_ * INNER_ * 4);   // 16 MB
  float*    kv_f32= (float*)   alloc((size_t)ROWS_ * 128 * 4);      //  4 MB
  _Float16* qn_h  = (_Float16*)alloc((size_t)ROWS_ * INNER_ * 2);   //  8 MB
  _Float16* kn_h  = (_Float16*)alloc((size_t)ROWS_ * DH_ * 2);      //  1 MB
  _Float16* vt_h  = (_Float16*)alloc((size_t)ROWS_ * DH_ * 2);      //  1 MB
  _Float16* ao_h  = (_Float16*)alloc((size_t)ROWS_ * INNER_ * 2);   //  8 MB

  // 1. weight transposes (f32 -> f16 [N][K])
  wt_kernel<<<(D_ * INNER_ + 255) / 256, 256, 0, stream>>>(Wq,   WqT,   D_,    INNER_);
  wt_kernel<<<(D_ * 128    + 255) / 256, 256, 0, stream>>>(Wkv,  WkvT,  D_,    128);
  wt_kernel<<<(INNER_ * D_ + 255) / 256, 256, 0, stream>>>(Wout, WoutT, INNER_, D_);

  // 2. LayerNorm(x) -> xn (f16)
  ln_to_h_kernel<<<ROWS_ / 4, 128, 0, stream>>>(x, norm_g, xn_h);

  // 3. q = xn@Wq ; kv = xn@Wkv   (32x64 tile per wave)
  gemm_kernel<<<dim3(INNER_ / 64, ROWS_ / 32), 32, 0, stream>>>(xn_h, WqT,  q_f32,  ROWS_, INNER_, D_);
  gemm_kernel<<<dim3(128    / 64, ROWS_ / 32), 32, 0, stream>>>(xn_h, WkvT, kv_f32, ROWS_, 128,    D_);

  // 4. l2norms + V transpose
  norm_q_kernel <<<(ROWS_ * H_) / 4, 128, 0, stream>>>(q_f32, qn_h);
  norm_kv_kernel<<<ROWS_ / 4,        128, 0, stream>>>(kv_f32, kn_h, vt_h);

  // 5. attention (async-LDS staged K/V, WMMA everywhere)
  attn_kernel<<<(B_ * H_ * (N_ / 16)) / 4, 128, 0, stream>>>(qn_h, kn_h, vt_h, ao_h);

  // 6. out = ao @ Wout  (f32 into d_out)
  gemm_kernel<<<dim3(D_ / 64, ROWS_ / 32), 32, 0, stream>>>(ao_h, WoutT, out, ROWS_, D_, INNER_);

  // 7. final LayerNorm in place on d_out
  ln_inplace_kernel<<<ROWS_ / 4, 128, 0, stream>>>(out, out_norm_g);
}
